// SparseSelfAttention_13941463843035
// MI455X (gfx1250) — compile-verified, matrix operand order checked by reading, not run on hardware
//
#include <hip/hip_runtime.h>
#include <hip/hip_bf16.h>

// ---------------------------------------------------------------------------
// SparseSelfAttention for MI455X (gfx1250, wave32, WMMA bf16 16x16x32)
//
//   x:      (B=8, C=640, T=2048) f32
//   w_qkv:  (1920, 640) f32      b_qkv: (1920,) f32
//   w_out:  (640, 640)  f32      b_out: (640,)  f32
//   mask:   (2048, 2048) bool
//   out:    (8, 640, 2048) f32
//
// Pipeline:
//   k0: weights -> bf16
//   k1: x -> XT[b][t][c] bf16 (LDS tile transpose; makes GEMM B-frags
//       contiguous; 21MB => L2-resident on 192MB L2)
//   k2: QKV GEMM, pure-register WMMA (L2 feeds both fragments)
//   k3: flash attention: block-shared K/V tiles double-buffered via
//       global_load_async_to_lds_b128, WMMA scores + online softmax +
//       per-wave LDS P-transpose + WMMA P@V
//   k4: output projection, pure-register WMMA
// ---------------------------------------------------------------------------

#define DIMC   640
#define HEADS  5
#define HD     128
#define BB     8
#define TT     2048
#define ODIM   1920   // 3*DIMC
#define WIN    5      // WINDOW//2

typedef __attribute__((ext_vector_type(16))) __bf16 bf16x16;
typedef __attribute__((ext_vector_type(8)))  __bf16 bf16x8;
typedef __attribute__((ext_vector_type(8)))  float  f32x8;

// Build one 16-bit A/B WMMA fragment from two contiguous 16B chunks.
// Per ISA 16-bit layout: lane half h holds K = [8h, 8h+8) in elems 0..7 and
// K = [8h+16, 8h+24) in elems 8..15 -> two aligned 16-byte loads.
static __device__ __forceinline__ bf16x16 frag2(const __bf16* p0, const __bf16* p1) {
    bf16x8 lo = *reinterpret_cast<const bf16x8*>(p0);
    bf16x8 hi = *reinterpret_cast<const bf16x8*>(p1);
    return __builtin_shufflevector(lo, hi, 0,1,2,3,4,5,6,7,8,9,10,11,12,13,14,15);
}

static __device__ __forceinline__ f32x8 wmma_bf16(bf16x16 a, bf16x16 b, f32x8 c) {
    return __builtin_amdgcn_wmma_f32_16x16x32_bf16(false, a, false, b, (short)0, c,
                                                   false, false);
}

// gfx1250 async global->LDS copy (16B per lane issue), tracked by ASYNCcnt.
// Generic shared pointer low 32 bits == LDS byte offset (aperture form).
static __device__ __forceinline__ void async_ld16(const __bf16* lds_dst,
                                                  const __bf16* gsrc) {
    unsigned int       loff = (unsigned int)(unsigned long long)lds_dst;
    unsigned long long gp   = (unsigned long long)gsrc;
    asm volatile("global_load_async_to_lds_b128 %0, %1, off"
                 :: "v"(loff), "v"(gp) : "memory");
}

static __device__ __forceinline__ void wait_async0() {
    asm volatile("s_wait_asynccnt 0" ::: "memory");
}

// ---------------------------------------------------------------------------
// Kernel 0: f32 -> bf16 conversion (weights)
// ---------------------------------------------------------------------------
__global__ void cvt_bf16_kernel(const float* __restrict__ src,
                                __bf16* __restrict__ dst, int n) {
    for (int i = blockIdx.x * blockDim.x + threadIdx.x; i < n;
         i += gridDim.x * blockDim.x) {
        dst[i] = (__bf16)src[i];
    }
}

// ---------------------------------------------------------------------------
// Kernel 1: x (B,C,T) f32 -> XT (B,T,C) bf16, LDS 32x33 tile transpose
// ---------------------------------------------------------------------------
__global__ void xpose_kernel(const float* __restrict__ x,
                             __bf16* __restrict__ XT) {
    __shared__ float tile[32][33];
    const int t0 = blockIdx.x * 32;
    const int c0 = blockIdx.y * 32;
    const int b  = blockIdx.z;
    const int tx = threadIdx.x;      // 0..31
    const int ty = threadIdx.y;      // 0..7
    #pragma unroll
    for (int i = 0; i < 32; i += 8)
        tile[ty + i][tx] = x[((size_t)b * DIMC + c0 + ty + i) * TT + t0 + tx];
    __syncthreads();
    #pragma unroll
    for (int i = 0; i < 32; i += 8)
        XT[((size_t)b * TT + t0 + ty + i) * DIMC + c0 + tx] = (__bf16)tile[tx][ty + i];
}

// ---------------------------------------------------------------------------
// Kernel 2: QKV projection, pure-register WMMA GEMM.
// Wave tile: 16 (o) x 32 (t); A fragment reused across both N halves.
// Epilogue scatters into Q,K (t,d-major) and V^T (d,t) in bf16.
// ---------------------------------------------------------------------------
__global__ void qkv_gemm_kernel(const __bf16* __restrict__ XT,    // (B,T,640)
                                const __bf16* __restrict__ w,     // (1920,640)
                                const float* __restrict__ bias,   // (1920,)
                                __bf16* __restrict__ Q,           // (B,H,T,HD)
                                __bf16* __restrict__ K,           // (B,H,T,HD)
                                __bf16* __restrict__ VT) {        // (B,H,HD,T)
    const int wv    = threadIdx.x >> 5;
    const int lane  = threadIdx.x & 31;
    const int l16   = lane & 15;
    const int lhalf = lane >> 4;

    const int wid = blockIdx.x * 4 + wv;     // B * 120 * 64 waves
    const int t2  = wid & 63;                // t tile (32 wide)
    const int ot  = (wid >> 6) % 120;        // o tile (16 tall)
    const int b   = wid / (64 * 120);
    const int o0  = ot << 4;
    const int t0  = t2 << 5;

    f32x8 acc0 = {}, acc1 = {};
    #pragma unroll 4
    for (int k0 = 0; k0 < DIMC; k0 += 32) {
        const __bf16* ap = w + (size_t)(o0 + l16) * DIMC + k0 + (lhalf << 3);
        bf16x16 a = frag2(ap, ap + 16);
        const __bf16* bp0 = XT + ((size_t)b * TT + t0 + l16) * DIMC + k0 + (lhalf << 3);
        const __bf16* bp1 = bp0 + (size_t)16 * DIMC;
        acc0 = wmma_bf16(a, frag2(bp0, bp0 + 16), acc0);
        acc1 = wmma_bf16(a, frag2(bp1, bp1 + 16), acc1);
    }

    // Epilogue: lane holds rows o = o0 + r + 8*lhalf, cols t0+l16 / t0+16+l16.
    const int sel = o0 / DIMC;               // o tile is 16-aligned: sel uniform
    const int rem = o0 % DIMC;
    const int h   = rem >> 7;
    const size_t bh = (size_t)b * HEADS + h;
    const int tg0 = t0 + l16, tg1 = t0 + 16 + l16;

    #pragma unroll
    for (int r = 0; r < 8; ++r) {
        const int o = o0 + r + (lhalf << 3);
        const int d = (o % DIMC) & 127;
        const float bo = bias[o];
        const __bf16 v0 = (__bf16)(acc0[r] + bo);
        const __bf16 v1 = (__bf16)(acc1[r] + bo);
        if (sel == 0) {
            Q[(bh * TT + tg0) * HD + d] = v0;
            Q[(bh * TT + tg1) * HD + d] = v1;
        } else if (sel == 1) {
            K[(bh * TT + tg0) * HD + d] = v0;
            K[(bh * TT + tg1) * HD + d] = v1;
        } else {
            VT[(bh * HD + d) * TT + tg0] = v0;
            VT[(bh * HD + d) * TT + tg1] = v1;
        }
    }
}

// ---------------------------------------------------------------------------
// Kernel 3: flash attention.
// Block = 4 waves = 4 consecutive 16-row q tiles of the SAME (b,h): K/V tiles
// are staged into LDS once per block via async copies, double-buffered.
// Per 32-key step: 8 WMMAs (scores) + online softmax + per-wave LDS P
// transpose + 8 WMMAs (P@V).
// ---------------------------------------------------------------------------
__global__ void attention_kernel(const __bf16* __restrict__ Q,
                                 const __bf16* __restrict__ K,
                                 const __bf16* __restrict__ VT,
                                 const unsigned char* __restrict__ mask,
                                 __bf16* __restrict__ OT) {
    // K tile rows padded to 136 halves (272B: 16-lane reads conflict-free),
    // V tile rows padded to 40 halves (80B).
    __shared__ __align__(16) __bf16 Kt[2][32][136];   // [key][d]
    __shared__ __align__(16) __bf16 Vt[2][128][40];   // [d][key]
    __shared__ __align__(16) __bf16 Pt[4][16][40];    // per-wave P (16q x 32k)

    const int tid   = threadIdx.x;
    const int wv    = tid >> 5;
    const int lane  = tid & 31;
    const int l16   = lane & 15;
    const int lhalf = lane >> 4;

    const int wid = blockIdx.x * 4 + wv;            // 0 .. B*H*(T/16)-1
    const int qt  = wid & 127;                      // T/16 = 128
    const int h   = (wid >> 7) % HEADS;
    const int b   = wid / (128 * HEADS);
    const int q0  = qt << 4;

    const size_t bh = (size_t)b * HEADS + h;
    const __bf16* Qb  = Q  + bh * TT * HD;
    const __bf16* Kb  = K  + bh * TT * HD;
    const __bf16* VTb = VT + bh * HD * TT;
    __bf16 (*pl)[40] = Pt[wv];

    // Preload Q fragments (row = q0+l16, 4 K-chunks of 32 over head dim)
    bf16x16 qfrag[4];
    #pragma unroll
    for (int i = 0; i < 4; ++i) {
        const __bf16* qp = Qb + (size_t)(q0 + l16) * HD + i * 32 + (lhalf << 3);
        qfrag[i] = frag2(qp, qp + 16);
    }

    const float scale = 0.08838834764831845f;  // 1/sqrt(128)
    float rm[8], rs[8];
    f32x8 accv[8];
    #pragma unroll
    for (int r = 0; r < 8; ++r) { rm[r] = -3.0e38f; rs[r] = 0.0f; }
    #pragma unroll
    for (int c = 0; c < 8; ++c) accv[c] = f32x8{};

    const int qrowbase = q0 + (lhalf << 3);

    // ---- prologue: async-stage tile 0 into buffer 0 ----
    #pragma unroll
    for (int i = 0; i < 4; ++i) {
        const int c  = tid + 128 * i;                 // 512 16B chunks each
        const int kr = c >> 4, kc = (c & 15) << 3;    // K: 32 rows x 16 chunks
        async_ld16(&Kt[0][kr][kc], Kb + (size_t)kr * HD + kc);
        const int vr = c >> 2, vc = (c & 3) << 3;     // V: 128 rows x 4 chunks
        async_ld16(&Vt[0][vr][vc], VTb + (size_t)vr * TT + vc);
    }
    wait_async0();
    __syncthreads();

    int buf = 0;
    for (int k0 = 0; k0 < TT; k0 += 32) {
        // ---- async-prefetch next K/V tile into the other buffer ----
        if (k0 + 32 < TT) {
            const int kn = k0 + 32;
            #pragma unroll
            for (int i = 0; i < 4; ++i) {
                const int c  = tid + 128 * i;
                const int kr = c >> 4, kc = (c & 15) << 3;
                async_ld16(&Kt[buf ^ 1][kr][kc], Kb + (size_t)(kn + kr) * HD + kc);
                const int vr = c >> 2, vc = (c & 3) << 3;
                async_ld16(&Vt[buf ^ 1][vr][vc], VTb + (size_t)vr * TT + kn + vc);
            }
        }

        // ---- scores: two 16x16 C tiles over 32 keys, K=128 chained ----
        f32x8 s0 = {}, s1 = {};
        #pragma unroll
        for (int i = 0; i < 4; ++i) {
            const __bf16* kp = &Kt[buf][l16][i * 32 + (lhalf << 3)];
            s0 = wmma_bf16(qfrag[i], frag2(kp, kp + 16), s0);
        }
        #pragma unroll
        for (int i = 0; i < 4; ++i) {
            const __bf16* kp = &Kt[buf][16 + l16][i * 32 + (lhalf << 3)];
            s1 = wmma_bf16(qfrag[i], frag2(kp, kp + 16), s1);
        }

        // ---- scale + sparse/window bias ----
        float tmax[8];
        #pragma unroll
        for (int r = 0; r < 8; ++r) {
            const int qr  = qrowbase + r;
            const int kc0 = k0 + l16;
            const int kc1 = kc0 + 16;
            const int d0  = qr - kc0, d1 = qr - kc1;
            const bool a0 = ((unsigned)(d0 + WIN) <= 2u * WIN) ||
                            mask[(size_t)qr * TT + kc0];
            const bool a1 = ((unsigned)(d1 + WIN) <= 2u * WIN) ||
                            mask[(size_t)qr * TT + kc1];
            const float v0 = s0[r] * scale + (a0 ? 1.0f : -1.0e9f);
            const float v1 = s1[r] * scale + (a1 ? 1.0f : -1.0e9f);
            s0[r] = v0; s1[r] = v1;
            tmax[r] = fmaxf(v0, v1);
        }
        // row-max over the 16-lane N group
        #pragma unroll
        for (int r = 0; r < 8; ++r)
            #pragma unroll
            for (int m = 1; m < 16; m <<= 1)
                tmax[r] = fmaxf(tmax[r], __shfl_xor(tmax[r], m, 16));

        // ---- online softmax update + write P to LDS (C->A transpose) ----
        float alpha[8], tsum[8];
        #pragma unroll
        for (int r = 0; r < 8; ++r) {
            const float mnew = fmaxf(rm[r], tmax[r]);
            alpha[r] = __expf(rm[r] - mnew);
            rm[r] = mnew;
            const float e0 = __expf(s0[r] - mnew);
            const float e1 = __expf(s1[r] - mnew);
            pl[r + (lhalf << 3)][l16]      = (__bf16)e0;
            pl[r + (lhalf << 3)][l16 + 16] = (__bf16)e1;
            tsum[r] = e0 + e1;
        }
        #pragma unroll
        for (int r = 0; r < 8; ++r)
            #pragma unroll
            for (int m = 1; m < 16; m <<= 1)
                tsum[r] += __shfl_xor(tsum[r], m, 16);
        #pragma unroll
        for (int r = 0; r < 8; ++r) rs[r] = rs[r] * alpha[r] + tsum[r];

        // rescale running output accumulators
        #pragma unroll
        for (int c = 0; c < 8; ++c)
            #pragma unroll
            for (int r = 0; r < 8; ++r)
                accv[c][r] *= alpha[r];

        // wave-local LDS fence for the P transpose (same-wave producer/consumer)
        asm volatile("s_wait_dscnt 0" ::: "memory");

        // ---- P @ V : A = P (16x32), B = V tile rows (d-major, key contig) ----
        const __bf16* pp = &pl[l16][lhalf << 3];
        bf16x16 pf = frag2(pp, pp + 16);
        #pragma unroll
        for (int c = 0; c < 8; ++c) {
            const __bf16* vp = &Vt[buf][c * 16 + l16][lhalf << 3];
            accv[c] = wmma_bf16(pf, frag2(vp, vp + 16), accv[c]);
        }

        // ---- rotate double buffer ----
        wait_async0();
        __syncthreads();
        buf ^= 1;
    }

    // ---- normalize + store to OT[b][t][h*128+d] (t-major, c contiguous) ----
    __bf16* OTb = OT + (size_t)b * TT * DIMC;
    #pragma unroll
    for (int c = 0; c < 8; ++c) {
        const int d = h * HD + c * 16 + l16;
        #pragma unroll
        for (int r = 0; r < 8; ++r) {
            const int t = qrowbase + r;
            OTb[(size_t)t * DIMC + d] = (__bf16)(accv[c][r] / rs[r]);
        }
    }
}

// ---------------------------------------------------------------------------
// Kernel 4: output projection  y[b,o,t] = sum_c wout[o,c]*OT[b,t,c] + bout[o]
// Pure-register WMMA GEMM; wave tile 16 (o) x 32 (t), A fragment reused.
// ---------------------------------------------------------------------------
__global__ void out_proj_kernel(const __bf16* __restrict__ wout,  // (640,640)
                                const __bf16* __restrict__ OT,    // (B,T,640)
                                const float* __restrict__ bout,
                                float* __restrict__ out) {        // (B,640,T)
    const int wv    = threadIdx.x >> 5;
    const int lane  = threadIdx.x & 31;
    const int l16   = lane & 15;
    const int lhalf = lane >> 4;

    const int wid = blockIdx.x * 4 + wv;     // B * 40 * 64 waves
    const int t2  = wid & 63;                // t tile (32 wide)
    const int ot  = (wid >> 6) % 40;         // o tile (16 tall)
    const int b   = wid / (64 * 40);
    const int o0  = ot << 4;
    const int t0  = t2 << 5;

    f32x8 acc0 = {}, acc1 = {};
    #pragma unroll 4
    for (int k0 = 0; k0 < DIMC; k0 += 32) {
        const __bf16* ap = wout + (size_t)(o0 + l16) * DIMC + k0 + (lhalf << 3);
        bf16x16 a = frag2(ap, ap + 16);
        const __bf16* bp0 = OT + ((size_t)b * TT + t0 + l16) * DIMC + k0 + (lhalf << 3);
        const __bf16* bp1 = bp0 + (size_t)16 * DIMC;
        acc0 = wmma_bf16(a, frag2(bp0, bp0 + 16), acc0);
        acc1 = wmma_bf16(a, frag2(bp1, bp1 + 16), acc1);
    }

    #pragma unroll
    for (int r = 0; r < 8; ++r) {
        const int o = o0 + (lhalf << 3) + r;
        const float bo = bout[o];
        out[((size_t)b * DIMC + o) * TT + t0 + l16]      = acc0[r] + bo;
        out[((size_t)b * DIMC + o) * TT + t0 + 16 + l16] = acc1[r] + bo;
    }
}

// ---------------------------------------------------------------------------
// Launch
// ---------------------------------------------------------------------------
extern "C" void kernel_launch(void* const* d_in, const int* in_sizes, int n_in,
                              void* d_out, int out_size, void* d_ws, size_t ws_size,
                              hipStream_t stream) {
    const float*         x      = (const float*)d_in[0];
    const float*         w_qkv  = (const float*)d_in[1];
    const float*         b_qkv  = (const float*)d_in[2];
    const float*         w_out  = (const float*)d_in[3];
    const float*         b_out  = (const float*)d_in[4];
    const unsigned char* mask   = (const unsigned char*)d_in[5];
    float*               out    = (float*)d_out;

    char* ws = (char*)d_ws;
    // bf16 workspace layout (bytes, 256-aligned)
    __bf16* wqkv_h = (__bf16*)(ws);                 //  1920*640*2 =  2,457,600
    __bf16* wout_h = (__bf16*)(ws +   2457600);     //   640*640*2 =    819,200
    __bf16* XTws   = (__bf16*)(ws +   3276800);     // B*T*640*2   = 20,971,520
    __bf16* Qws    = (__bf16*)(ws +  24248320);     // 20,971,520
    __bf16* Kws    = (__bf16*)(ws +  45219840);     // 20,971,520
    __bf16* VTws   = (__bf16*)(ws +  66191360);     // 20,971,520
    __bf16* OTws   = (__bf16*)(ws +  87162880);     // 20,971,520
    // total: 108,134,400 bytes

    cvt_bf16_kernel<<<dim3(1200), 256, 0, stream>>>(w_qkv, wqkv_h, ODIM * DIMC);
    cvt_bf16_kernel<<<dim3(400),  256, 0, stream>>>(w_out, wout_h, DIMC * DIMC);

    // x transpose+convert: tiles (T/32, C/32, B)
    xpose_kernel<<<dim3(64, 20, BB), dim3(32, 8), 0, stream>>>(x, XTws);

    // QKV GEMM: B*120*64 wave-tiles / 4 per block
    qkv_gemm_kernel<<<dim3(15360), 128, 0, stream>>>(XTws, wqkv_h, b_qkv,
                                                     Qws, Kws, VTws);

    // attention: B*H*(T/16)=5120 waves / 4 per block (4 q-tiles share K/V tiles)
    attention_kernel<<<dim3(1280), 128, 0, stream>>>(Qws, Kws, VTws, mask, OTws);

    // output projection: B*40*64 = 20480 wave-tiles / 4 per block
    out_proj_kernel<<<dim3(5120), 128, 0, stream>>>(wout_h, OTws, b_out, out);
}
